// Discriminator_31164282700385
// MI455X (gfx1250) — matrix operand order repaired and checked by reading, not compiled, not run on hardware
//
#include <hip/hip_runtime.h>
#include <hip/hip_bf16.h>
#include <math.h>

typedef __attribute__((ext_vector_type(16))) _Float16 v16h;
typedef __attribute__((ext_vector_type(8)))  _Float16 v8h;
typedef __attribute__((ext_vector_type(8)))  float    v8f;

#define BATCH  4096
#define TSTEPS 16
#define PREV   6
#define PRED   10
#define FEAT   64
#define HID    256
#define GATES  1024   // 4*HID

#define BM 32         // batch rows per workgroup
#define ZS 1032       // padded z row stride (floats)
#define HS 272        // padded h row stride (halves), 16B-aligned rows

#define ZBYTES (BM * ZS * 4)
#define HBYTES (BM * HS * 2)
#define PBYTES (BM * 8 * 4)
#define SMEM_BYTES (ZBYTES + HBYTES + PBYTES)

union V16 { v16h v; v8h h[2]; };

__device__ __forceinline__ float sigmoidf_(float x) { return 1.0f / (1.0f + __expf(-x)); }

// ---------------- prep: fp32 -> f16 weight transpose ----------------
__global__ void prep_weights(const float* __restrict__ K, const float* __restrict__ R,
                             _Float16* __restrict__ KT, _Float16* __restrict__ RT) {
    const int i = blockIdx.x * blockDim.x + threadIdx.x;
    if (i < GATES * FEAT) {                 // KT[n][k] = K[k][n]
        const int n = i / FEAT, k = i % FEAT;
        KT[i] = (_Float16)K[(size_t)k * GATES + n];
    }
    if (i < GATES * HID) {                  // RT[n][k] = R[k][n]
        const int n = i / HID, k = i % HID;
        RT[i] = (_Float16)R[(size_t)k * GATES + n];
    }
}

// ---------------- prep: fp32 -> f16 inputs, build fake sequence ----------------
__global__ void prep_inputs(const float* __restrict__ real, const float* __restrict__ fake,
                            _Float16* __restrict__ xr, _Float16* __restrict__ xf) {
    const int i = blockIdx.x * blockDim.x + threadIdx.x;
    if (i < BATCH * TSTEPS * FEAT) {
        const int f = i & (FEAT - 1);
        const int t = (i >> 6) & (TSTEPS - 1);
        const int b = i >> 10;
        xr[i] = (_Float16)real[i];
        const float v = (t < PREV) ? real[i]
                                   : fake[((size_t)b * PRED + (t - PREV)) * FEAT + f];
        xf[i] = (_Float16)v;
    }
}

// ---------------- main: persistent-state LSTM + head ----------------
__global__ __launch_bounds__(256) void lstm_disc_kernel(
    const _Float16* __restrict__ xr, const _Float16* __restrict__ xf,
    const _Float16* __restrict__ KT, const _Float16* __restrict__ RT,
    const float* __restrict__ bias, const float* __restrict__ dw,
    const float* __restrict__ db, float* __restrict__ out)
{
    extern __shared__ char smem[];
    float*    zbuf = (float*)smem;                      // [BM][ZS]
    _Float16* hbuf = (_Float16*)(smem + ZBYTES);        // [BM][HS]
    float*    pbuf = (float*)(smem + ZBYTES + HBYTES);  // [BM][8]

    const int seq   = blockIdx.y;
    const int b0    = blockIdx.x * BM;
    const int tid   = threadIdx.x;
    const int wave  = tid >> 5;
    const int lane  = tid & 31;
    const int mrow  = lane & 15;      // A row / B,C column within a 16-tile
    const int khalf = lane >> 4;      // 0 or 1
    const int kb    = khalf * 8;      // A-operand K sub-offset
    const int n0    = wave * 128;     // this wave's gate-column base

    const _Float16* xp = (seq == 0) ? xr : xf;

    // per-column bias (accumulator init)
    float bb[8];
    #pragma unroll
    for (int ct = 0; ct < 8; ++ct) bb[ct] = bias[n0 + ct * 16 + mrow];

    // elementwise-phase ownership: row er, columns ec0..ec0+31
    const int er  = tid >> 3;
    const int ec0 = (tid & 7) * 32;
    float wreg[32];
    #pragma unroll
    for (int k = 0; k < 32; ++k) wreg[k] = dw[ec0 + k];
    float creg[32];
    #pragma unroll
    for (int k = 0; k < 32; ++k) creg[k] = 0.0f;
    const float dbv = db[0];

    #pragma unroll 1
    for (int t = 0; t < TSTEPS; ++t) {
        v8f acc[2][8];
        #pragma unroll
        for (int rt = 0; rt < 2; ++rt)
            #pragma unroll
            for (int ct = 0; ct < 8; ++ct)
                #pragma unroll
                for (int v = 0; v < 8; ++v)
                    acc[rt][ct][v] = bb[ct];

        // ---- x_t @ K : K-dim 64 -> 2 chunks of 32
        #pragma unroll
        for (int kc = 0; kc < 2; ++kc) {
            V16 a[2];
            #pragma unroll
            for (int rt = 0; rt < 2; ++rt) {
                const _Float16* p = xp + ((size_t)(b0 + rt * 16 + mrow) * TSTEPS + t) * FEAT
                                       + kc * 32 + kb;
                a[rt].h[0] = *(const v8h*)p;
                a[rt].h[1] = *(const v8h*)(p + 16);
            }
            #pragma unroll
            for (int ct = 0; ct < 8; ++ct) {
                V16 bm;
                const _Float16* bp = KT + (size_t)(n0 + ct * 16 + mrow) * FEAT
                                        + kc * 32 + kb;
                bm.h[0] = *(const v8h*)bp;
                bm.h[1] = *(const v8h*)(bp + 16);
                acc[0][ct] = __builtin_amdgcn_wmma_f32_16x16x32_f16(
                    false, a[0].v, false, bm.v, (short)0, acc[0][ct], false, false);
                acc[1][ct] = __builtin_amdgcn_wmma_f32_16x16x32_f16(
                    false, a[1].v, false, bm.v, (short)0, acc[1][ct], false, false);
            }
        }

        // ---- h_{t-1} @ R : K-dim 256 -> 8 chunks of 32 (h == 0 at t == 0)
        if (t > 0) {
            #pragma unroll
            for (int kc = 0; kc < 8; ++kc) {
                V16 a[2];
                #pragma unroll
                for (int rt = 0; rt < 2; ++rt) {
                    const _Float16* p = hbuf + (rt * 16 + mrow) * HS + kc * 32 + kb;
                    a[rt].h[0] = *(const v8h*)p;
                    a[rt].h[1] = *(const v8h*)(p + 16);
                }
                #pragma unroll
                for (int ct = 0; ct < 8; ++ct) {
                    V16 bm;
                    const _Float16* bp = RT + (size_t)(n0 + ct * 16 + mrow) * HID
                                            + kc * 32 + kb;
                    bm.h[0] = *(const v8h*)bp;
                    bm.h[1] = *(const v8h*)(bp + 16);
                    acc[0][ct] = __builtin_amdgcn_wmma_f32_16x16x32_f16(
                        false, a[0].v, false, bm.v, (short)0, acc[0][ct], false, false);
                    acc[1][ct] = __builtin_amdgcn_wmma_f32_16x16x32_f16(
                        false, a[1].v, false, bm.v, (short)0, acc[1][ct], false, false);
                }
            }
        }

        // ---- spill z tiles to LDS (C layout: VGPR v -> row v (+8 for hi lanes))
        #pragma unroll
        for (int rt = 0; rt < 2; ++rt)
            #pragma unroll
            for (int ct = 0; ct < 8; ++ct) {
                const int col = n0 + ct * 16 + mrow;
                #pragma unroll
                for (int v = 0; v < 8; ++v)
                    zbuf[(rt * 16 + khalf * 8 + v) * ZS + col] = acc[rt][ct][v];
            }
        __syncthreads();

        // ---- gates / state update / head partial dot
        float pdot = 0.0f;
        #pragma unroll
        for (int k = 0; k < 32; ++k) {
            const int j = ec0 + k;
            const float iz = zbuf[er * ZS + j];
            const float fz = zbuf[er * ZS + j + 256];
            const float gz = zbuf[er * ZS + j + 512];
            const float oz = zbuf[er * ZS + j + 768];
            const float ig = sigmoidf_(iz);
            const float fg = sigmoidf_(fz);
            const float gg = tanhf(gz);
            const float og = sigmoidf_(oz);
            const float c  = fg * creg[k] + ig * gg;
            creg[k] = c;
            const float h  = og * tanhf(c);
            hbuf[er * HS + j] = (_Float16)h;
            pdot += h * wreg[k];
        }
        pbuf[er * 8 + (tid & 7)] = pdot;
        __syncthreads();

        if (t >= PREV && tid < BM) {
            float s = dbv;
            #pragma unroll
            for (int q = 0; q < 8; ++q) s += pbuf[tid * 8 + q];
            out[(size_t)seq * BATCH * PRED + (size_t)(b0 + tid) * PRED + (t - PREV)]
                = sigmoidf_(s);
        }
    }
}

extern "C" void kernel_launch(void* const* d_in, const int* in_sizes, int n_in,
                              void* d_out, int out_size, void* d_ws, size_t ws_size,
                              hipStream_t stream) {
    const float* real = (const float*)d_in[0];
    const float* fake = (const float*)d_in[1];
    const float* kern = (const float*)d_in[2];
    const float* rker = (const float*)d_in[3];
    const float* bias = (const float*)d_in[4];
    const float* dw   = (const float*)d_in[5];
    const float* db   = (const float*)d_in[6];
    float* out = (float*)d_out;

    char* ws = (char*)d_ws;
    _Float16* KT = (_Float16*)ws;                                        // 128 KB
    _Float16* RT = (_Float16*)(ws + (size_t)GATES * FEAT * 2);           // 512 KB
    _Float16* xr = (_Float16*)(ws + (size_t)GATES * (FEAT + HID) * 2);   // 8 MB
    _Float16* xf = xr + (size_t)BATCH * TSTEPS * FEAT;                   // 8 MB

    prep_weights<<<(GATES * HID + 255) / 256, 256, 0, stream>>>(kern, rker, KT, RT);
    prep_inputs<<<(BATCH * TSTEPS * FEAT + 255) / 256, 256, 0, stream>>>(real, fake, xr, xf);

    dim3 grid(BATCH / BM, 2);
    lstm_disc_kernel<<<grid, 256, SMEM_BYTES, stream>>>(xr, xf, KT, RT, bias, dw, db, out);
}
